// GATActor_91036126806909
// MI455X (gfx1250) — compile-verified, matrix operand order checked by reading, not run on hardware
//
#include <hip/hip_runtime.h>
#include <hip/hip_bf16.h>

// ---------------- problem constants ----------------
#define NNODES   16384
#define NB       256
#define NPG      64
#define NEDGE    262144
#define NEDGE2   (NEDGE + NNODES)   // with self loops
#define NAG      5
#define NLIDAR   20
#define NLEMB    14
#define GNNIN    34                 // 20 feats + 14 lidar emb
#define KPAD1    64                 // GNNIN padded for WMMA K
#define FDIM     512                // HEADS*HID
#define NHEADS   8
#define HIDD     64
#define MAXKP    1024               // largest GEMM K (fc1)

typedef _Float16 v16h __attribute__((ext_vector_type(16)));
typedef _Float16 h8   __attribute__((ext_vector_type(8)));
typedef float    v8f  __attribute__((ext_vector_type(8)));

union FragU { v16h v; h8 h[2]; };

// monotone order-preserving float<->uint encoding for atomic max
__device__ __forceinline__ unsigned encf(float f) {
  unsigned u = __float_as_uint(f);
  return (u & 0x80000000u) ? ~u : (u | 0x80000000u);
}
__device__ __forceinline__ float decf(unsigned u) {
  return (u & 0x80000000u) ? __uint_as_float(u & 0x7fffffffu)
                           : __uint_as_float(~u);
}

// ---------------- lidar encoder + input build ----------------
__global__ void lidar_kernel(const float* __restrict__ x,
                             const float* __restrict__ c1w, const float* __restrict__ c1b,
                             const float* __restrict__ c2w, const float* __restrict__ c2b,
                             const float* __restrict__ encw, const float* __restrict__ encb,
                             const float* __restrict__ d1w, const float* __restrict__ d1b,
                             const float* __restrict__ d2w, const float* __restrict__ d2b,
                             _Float16* __restrict__ xg16,
                             float* __restrict__ outRaw, float* __restrict__ outRec) {
  int n = blockIdx.x * blockDim.x + threadIdx.x;
  if (n >= NNODES) return;
  const float* xr = x + (size_t)n * 40;
  float raw[NLIDAR];
#pragma unroll
  for (int t = 0; t < NLIDAR; ++t) { raw[t] = xr[20 + t]; outRaw[(size_t)n * NLIDAR + t] = raw[t]; }

  // conv1 (1->16, k5, pad2) + relu + maxpool2 -> pooled[16][10]
  float pooled[16][10];
  for (int c = 0; c < 16; ++c)
    for (int t2 = 0; t2 < 10; ++t2) {
      float mx = -1e30f;
#pragma unroll
      for (int half = 0; half < 2; ++half) {
        int t = t2 * 2 + half;
        float acc = c1b[c];
#pragma unroll
        for (int k = 0; k < 5; ++k) {
          int tt = t + k - 2;
          if (tt >= 0 && tt < NLIDAR) acc += c1w[c * 5 + k] * raw[tt];
        }
        mx = fmaxf(mx, fmaxf(acc, 0.f));
      }
      pooled[c][t2] = mx;
    }

  // conv2 (16->32, k3, pad1) + relu + mean over length -> feat[32]
  float feat[32];
  for (int o = 0; o < 32; ++o) {
    float sum = 0.f;
    for (int t = 0; t < 10; ++t) {
      float acc = c2b[o];
      for (int i = 0; i < 16; ++i)
#pragma unroll
        for (int k = 0; k < 3; ++k) {
          int tt = t + k - 1;
          if (tt >= 0 && tt < 10) acc += c2w[(o * 16 + i) * 3 + k] * pooled[i][tt];
        }
      sum += fmaxf(acc, 0.f);
    }
    feat[o] = sum * 0.1f;
  }

  float lid[NLEMB];
  for (int j = 0; j < NLEMB; ++j) {
    float a = encb[j];
    for (int i = 0; i < 32; ++i) a += feat[i] * encw[i * NLEMB + j];
    lid[j] = fmaxf(a, 0.f);
  }
  float d1[HIDD];
  for (int j = 0; j < HIDD; ++j) {
    float a = d1b[j];
    for (int i = 0; i < NLEMB; ++i) a += lid[i] * d1w[i * HIDD + j];
    d1[j] = fmaxf(a, 0.f);
  }
  for (int j = 0; j < NLIDAR; ++j) {
    float a = d2b[j];
    for (int i = 0; i < HIDD; ++i) a += d1[i] * d2w[i * NLIDAR + j];
    outRec[(size_t)n * NLIDAR + j] = a;
  }

  _Float16* xo = xg16 + (size_t)n * KPAD1;
  for (int j = 0; j < 20; ++j)          xo[j]      = (_Float16)xr[j];
  for (int j = 0; j < NLEMB; ++j)       xo[20 + j] = (_Float16)lid[j];
  for (int j = GNNIN; j < KPAD1; ++j)   xo[j]      = (_Float16)0.f;
}

// ---------------- f32 weight -> f16 transposed (N x Kpad) ----------------
__global__ void cvt_wt(const float* __restrict__ W, _Float16* __restrict__ Wt,
                       int K, int Nc, int Kp) {
  int idx = blockIdx.x * blockDim.x + threadIdx.x;
  if (idx >= Nc * Kp) return;
  int ncol = idx / Kp, k = idx - ncol * Kp;
  Wt[idx] = (k < K) ? (_Float16)W[(size_t)k * Nc + ncol] : (_Float16)0.f;
}

// ---------------- WMMA GEMM with async B-panel staging through LDS ----------------
// Block = 8 waves = 128x16 C macro-tile. The 16 x Kp B panel (weights for this
// N-tile) is DMA'd global->LDS once via GLOBAL_LOAD_ASYNC_TO_LDS_B128 (ASYNCcnt),
// then every wave reads B fragments with ds_load_b128 while streaming its own A
// rows from global. M must be a multiple of 128 (16384 and 1280 both are).
__global__ void gemm_wmma_lds(const _Float16* __restrict__ A, const _Float16* __restrict__ Bt,
                              const float* __restrict__ bias,
                              float* __restrict__ C, _Float16* __restrict__ C16,
                              int M, int Nc, int Kp, int relu) {
  __shared__ __align__(16) _Float16 Bsh[16 * MAXKP];
  const int mblocks = M >> 7;                  // blocks along M (128 rows each)
  const int bm = blockIdx.x % mblocks;
  const int tn = blockIdx.x / mblocks;
  const int wid  = threadIdx.x >> 5;
  const int lane = threadIdx.x & 31;

  // ---- async copy B panel (16 rows x Kp halfs, contiguous) into LDS ----
  const _Float16* Bpanel = Bt + (size_t)tn * 16 * Kp;
  const int chunks = (16 * Kp) >> 3;           // 16-byte chunks
  for (int c = threadIdx.x; c < chunks; c += 256) {
    unsigned ldsa = (unsigned)(unsigned long long)(uintptr_t)&Bsh[c * 8];
    const _Float16* gp = Bpanel + (size_t)c * 8;
    asm volatile("global_load_async_to_lds_b128 %0, %1, off"
                 :: "v"(ldsa), "v"(gp) : "memory");
  }
  asm volatile("s_wait_asynccnt 0x0" ::: "memory");
  __syncthreads();

  // ---- per-wave 16x16 tile ----
  const int tm    = bm * 8 + wid;
  const int rrow  = lane & 15;
  const int khalf = (lane >> 4) << 3;          // lanes 16..31 take K+8 slice (16b A layout)
  const _Float16* pa = A   + (size_t)(tm * 16 + rrow) * Kp + khalf;
  const _Float16* pb = Bsh + (size_t)rrow * Kp + khalf;

  v8f acc = {};
  for (int kb = 0; kb < Kp; kb += 32) {
    __builtin_prefetch((const void*)(pa + kb + 128), 0, 1);   // global_prefetch_b8
    FragU ua, ub;
    ua.h[0] = *(const h8*)(pa + kb);           // global_load_b128
    ua.h[1] = *(const h8*)(pa + kb + 16);
    ub.h[0] = *(const h8*)(pb + kb);           // ds_load_b128
    ub.h[1] = *(const h8*)(pb + kb + 16);
    acc = __builtin_amdgcn_wmma_f32_16x16x32_f16(
        /*neg_a=*/false, ua.v, /*neg_b=*/false, ub.v,
        /*c_mod=*/(short)0, acc, /*reuse_a=*/false, /*reuse_b=*/false);
  }

  const int cc = tn * 16 + (lane & 15);
  const float bv = bias ? bias[cc] : 0.f;
  const int rbase = tm * 16 + ((lane >> 4) << 3);
#pragma unroll
  for (int i = 0; i < 8; ++i) {
    float v = acc[i] + bv;
    if (relu) v = fmaxf(v, 0.f);
    size_t idx = (size_t)(rbase + i) * Nc + cc;
    C[idx] = v;
    if (C16) C16[idx] = (_Float16)v;
  }
}

// ---------------- attention: per (node, head) a_src / a_dst dot products ----------------
__global__ void asd_kernel(const float* __restrict__ h,
                           const float* __restrict__ att_src, const float* __restrict__ att_dst,
                           float* __restrict__ as, float* __restrict__ ad,
                           unsigned* __restrict__ amax, float* __restrict__ den) {
  int idx = blockIdx.x * blockDim.x + threadIdx.x;
  if (idx >= NNODES * NHEADS) return;
  int node = idx >> 3, hd = idx & 7;
  const float* hp = h + (size_t)node * FDIM + hd * HIDD;
  float s = 0.f, d = 0.f;
  for (int k = 0; k < HIDD; ++k) {
    s += hp[k] * att_src[hd * HIDD + k];
    d += hp[k] * att_dst[hd * HIDD + k];
  }
  as[idx] = s; ad[idx] = d;
  amax[idx] = 0u;        // == -inf sentinel
  den[idx]  = 0.f;
}

// ce[h] = sum_k W_e[0][h*64+k] * att_edge[h][k]   (edge_attr is scalar)
__global__ void ce_kernel(const float* __restrict__ We, const float* __restrict__ ae,
                          float* __restrict__ ce) {
  int hd = threadIdx.x;
  if (hd >= NHEADS) return;
  float s = 0.f;
  for (int k = 0; k < HIDD; ++k) s += We[hd * HIDD + k] * ae[hd * HIDD + k];
  ce[hd] = s;
}

// per-dst mean of edge_attr (for virtual self loops)
__global__ void emean_kernel(const int* __restrict__ dst, const float* __restrict__ eattr,
                             float* __restrict__ esum, float* __restrict__ ecnt) {
  int e = blockIdx.x * blockDim.x + threadIdx.x;
  if (e >= NEDGE) return;
  atomicAdd(&esum[dst[e]], eattr[e]);
  atomicAdd(&ecnt[dst[e]], 1.f);
}

// alpha = leaky_relu(a_s[src]+a_d[dst]+ea*ce, 0.2); atomic segment max
__global__ void alpha_kernel(const float* __restrict__ as, const float* __restrict__ ad,
                             const float* __restrict__ eattr, const float* __restrict__ ce,
                             const float* __restrict__ esum, const float* __restrict__ ecnt,
                             const int* __restrict__ src, const int* __restrict__ dst,
                             float* __restrict__ alpha, unsigned* __restrict__ amax,
                             int nE, int nTot) {
  int idx = blockIdx.x * blockDim.x + threadIdx.x;
  if (idx >= nTot * NHEADS) return;
  int e = idx >> 3, hd = idx & 7;
  int s, d; float ea;
  if (e < nE) { s = src[e]; d = dst[e]; ea = eattr[e]; }
  else        { s = d = e - nE; ea = esum[d] / fmaxf(ecnt[d], 1.f); }
  float a = as[s * NHEADS + hd] + ad[d * NHEADS + hd] + ea * ce[hd];
  a = (a > 0.f) ? a : 0.2f * a;
  alpha[idx] = a;
  atomicMax(&amax[d * NHEADS + hd], encf(a));
}

// ex = exp(alpha - amax[dst]); den += ex   (in-place on alpha buffer)
__global__ void exps_kernel(float* __restrict__ alpha, const unsigned* __restrict__ amax,
                            const int* __restrict__ dst, float* __restrict__ den,
                            int nE, int nTot) {
  int idx = blockIdx.x * blockDim.x + threadIdx.x;
  if (idx >= nTot * NHEADS) return;
  int e = idx >> 3, hd = idx & 7;
  int d = (e < nE) ? dst[e] : e - nE;
  unsigned mu = amax[d * NHEADS + hd];
  float m = (mu == 0u) ? 0.f : decf(mu);   // empty segment -> where(isfinite, ., 0)
  float ev = __expf(alpha[idx] - m);
  alpha[idx] = ev;
  atomicAdd(&den[d * NHEADS + hd], ev);
}

// out[n][c] = b[c]   (so aggregation lands on bias)
__global__ void binit_kernel(const float* __restrict__ b, float* __restrict__ out, int total) {
  int idx = blockIdx.x * blockDim.x + threadIdx.x;
  if (idx >= total) return;
  out[idx] = b[idx & (FDIM - 1)];
}

// wave-per-edge scatter: out[dst] += h[src] * coef   (L2-resident f32 atomics)
__global__ void aggregate_kernel(const float* __restrict__ h, const float* __restrict__ ex,
                                 const float* __restrict__ den,
                                 const int* __restrict__ src, const int* __restrict__ dst,
                                 float* __restrict__ out, int nE, int nTot) {
  int e = blockIdx.x * (blockDim.x >> 5) + (threadIdx.x >> 5);
  if (e >= nTot) return;
  int lane = threadIdx.x & 31;
  int s, d;
  if (e < nE) { s = src[e]; d = dst[e]; } else { s = d = e - nE; }
#pragma unroll
  for (int it = 0; it < FDIM / 32; ++it) {
    int c = lane + it * 32;
    int hd = c >> 6;
    float coef = ex[e * NHEADS + hd] / fmaxf(den[d * NHEADS + hd], 1e-16f);
    atomicAdd(&out[(size_t)d * FDIM + c], h[(size_t)s * FDIM + c] * coef);
  }
}

// relu in place + f16 re-quantize for next layer's A operand
__global__ void relu_cvt_kernel(float* __restrict__ x, _Float16* __restrict__ x16, int total) {
  int idx = blockIdx.x * blockDim.x + threadIdx.x;
  if (idx >= total) return;
  float v = fmaxf(x[idx], 0.f);
  x[idx] = v;
  x16[idx] = (_Float16)v;
}

// per-graph mean pooling (batch is contiguous: 64 nodes per graph)
__global__ void pool_kernel(const float* __restrict__ h3, float* __restrict__ gemb) {
  int idx = blockIdx.x * blockDim.x + threadIdx.x;
  if (idx >= NB * FDIM) return;
  int b = idx >> 9, c = idx & (FDIM - 1);
  float s = 0.f;
  for (int i = 0; i < NPG; ++i) s += h3[((size_t)(b * NPG + i)) * FDIM + c];
  gemb[idx] = s * (1.f / NPG);
}

// comb[row 1280][1024] = [agent h3 | gemb]  (f16 for fc1 WMMA)
__global__ void comb_kernel(const float* __restrict__ h3, const float* __restrict__ gemb,
                            _Float16* __restrict__ comb) {
  int idx = blockIdx.x * blockDim.x + threadIdx.x;
  if (idx >= NB * NAG * 1024) return;
  int row = idx >> 10, c = idx & 1023;
  int b = row / NAG, a = row - b * NAG;
  float v = (c < FDIM) ? h3[((size_t)(b * NPG + a)) * FDIM + c]
                       : gemb[b * FDIM + (c - FDIM)];
  comb[idx] = (_Float16)v;
}

// fc2 + tanh/sigmoid heads -> mean, std
__global__ void head_kernel(const float* __restrict__ hh, const float* __restrict__ w2,
                            const float* __restrict__ b2,
                            float* __restrict__ outMean, float* __restrict__ outStd) {
  int idx = blockIdx.x * blockDim.x + threadIdx.x;
  if (idx >= NB * NAG * 6) return;
  int row = idx / 6, j = idx - row * 6;
  float s = b2[j];
  for (int k = 0; k < 256; ++k) s += hh[(size_t)row * 256 + k] * w2[k * 6 + j];
  if (j < 3) {
    outMean[row * 3 + j] = tanhf(s) * 0.3f;           // LIMITS = 0.3
  } else {
    float sg = 1.f / (1.f + __expf(-s));
    outStd[row * 3 + (j - 3)] = 0.01f + sg * (0.3f - 0.01f) + 1e-5f;
  }
}

// ---------------- host side ----------------
static inline void* bump(char*& p, size_t bytes) {
  void* r = (void*)p;
  p += (bytes + 255) & ~(size_t)255;
  return r;
}

extern "C" void kernel_launch(void* const* d_in, const int* in_sizes, int n_in,
                              void* d_out, int out_size, void* d_ws, size_t ws_size,
                              hipStream_t stream) {
  // pytree leaves in insertion order:
  // 0:x 1:edge_attr 2..11: conv/enc/dec params, 12..17:g1, 18..23:g2, 24..29:g3,
  // 30:fc1_w 31:fc1_b 32:fc2_w 33:fc2_b 34:edge_index 35:batch
  const float* x     = (const float*)d_in[0];
  const float* eattr = (const float*)d_in[1];
  const float* c1w = (const float*)d_in[2];  const float* c1b = (const float*)d_in[3];
  const float* c2w = (const float*)d_in[4];  const float* c2b = (const float*)d_in[5];
  const float* encw= (const float*)d_in[6];  const float* encb= (const float*)d_in[7];
  const float* d1w = (const float*)d_in[8];  const float* d1b = (const float*)d_in[9];
  const float* d2w = (const float*)d_in[10]; const float* d2b = (const float*)d_in[11];
  const float* gW[3], *gAS[3], *gAD[3], *gWE[3], *gAE[3], *gB[3];
  for (int l = 0; l < 3; ++l) {
    int base = 12 + l * 6;
    gW[l]  = (const float*)d_in[base + 0];
    gAS[l] = (const float*)d_in[base + 1];
    gAD[l] = (const float*)d_in[base + 2];
    gWE[l] = (const float*)d_in[base + 3];
    gAE[l] = (const float*)d_in[base + 4];
    gB[l]  = (const float*)d_in[base + 5];
  }
  const float* fc1w = (const float*)d_in[30]; const float* fc1b = (const float*)d_in[31];
  const float* fc2w = (const float*)d_in[32]; const float* fc2b = (const float*)d_in[33];
  const int* eidx = (const int*)d_in[34];
  const int* srcI = eidx;
  const int* dstI = eidx + NEDGE;

  float* out_mean = (float*)d_out;                    // 256*5*3
  float* out_std  = out_mean + NB * NAG * 3;
  float* out_raw  = out_std  + NB * NAG * 3;          // 16384*20
  float* out_rec  = out_raw  + NNODES * NLIDAR;

  // ---- workspace carve-up ----
  char* wp = (char*)d_ws;
  _Float16* xg16   = (_Float16*)bump(wp, (size_t)NNODES * KPAD1 * 2);
  _Float16* wt16   = (_Float16*)bump(wp, (size_t)512 * 1024 * 2);     // max of all Bt
  float*    hbuf   = (float*)   bump(wp, (size_t)NNODES * FDIM * 4);  // pre-aggregation h
  float*    obuf   = (float*)   bump(wp, (size_t)NNODES * FDIM * 4);  // aggregated out / x
  _Float16* act16  = (_Float16*)bump(wp, (size_t)NNODES * FDIM * 2);
  float*    a_s    = (float*)   bump(wp, (size_t)NNODES * NHEADS * 4);
  float*    a_d    = (float*)   bump(wp, (size_t)NNODES * NHEADS * 4);
  unsigned* amax   = (unsigned*)bump(wp, (size_t)NNODES * NHEADS * 4);
  float*    den    = (float*)   bump(wp, (size_t)NNODES * NHEADS * 4);
  float*    exbuf  = (float*)   bump(wp, (size_t)NEDGE2 * NHEADS * 4);
  float*    esum   = (float*)   bump(wp, (size_t)NNODES * 4);
  float*    ecnt   = (float*)   bump(wp, (size_t)NNODES * 4);
  float*    cebuf  = (float*)   bump(wp, (size_t)NHEADS * 4);
  float*    gemb   = (float*)   bump(wp, (size_t)NB * FDIM * 4);
  _Float16* comb16 = (_Float16*)bump(wp, (size_t)NB * NAG * 1024 * 2);
  float*    hh     = (float*)   bump(wp, (size_t)NB * NAG * 256 * 4);
  (void)ws_size; (void)n_in; (void)in_sizes; (void)out_size;

  // ---- 1. lidar encoder / input build ----
  lidar_kernel<<<(NNODES + 63) / 64, 64, 0, stream>>>(
      x, c1w, c1b, c2w, c2b, encw, encb, d1w, d1b, d2w, d2b,
      xg16, out_raw, out_rec);

  // ---- 2. per-dst edge-attr mean for virtual self loops ----
  hipMemsetAsync(esum, 0, (size_t)NNODES * 4, stream);
  hipMemsetAsync(ecnt, 0, (size_t)NNODES * 4, stream);
  emean_kernel<<<(NEDGE + 255) / 256, 256, 0, stream>>>(dstI, eattr, esum, ecnt);

  const int totF = NNODES * FDIM;
  // ---- 3. three GAT layers ----
  for (int l = 0; l < 3; ++l) {
    const int Kp = (l == 0) ? KPAD1 : FDIM;
    const int Kr = (l == 0) ? GNNIN : FDIM;
    const _Float16* Ain = (l == 0) ? xg16 : act16;
    const int nE   = NEDGE;
    const int nTot = (l == 0) ? NEDGE : NEDGE2;

    // weights -> f16 transposed
    cvt_wt<<<(FDIM * Kp + 255) / 256, 256, 0, stream>>>(gW[l], wt16, Kr, FDIM, Kp);

    // h = x @ W   (WMMA, B panel staged via async global->LDS)
    {
      int blocks = (NNODES >> 7) * (FDIM >> 4);   // 128 * 32
      gemm_wmma_lds<<<blocks, 256, 0, stream>>>(
          Ain, wt16, nullptr, hbuf, nullptr, NNODES, FDIM, Kp, 0);
    }

    // attention scalars + segment softmax
    asd_kernel<<<(NNODES * NHEADS + 255) / 256, 256, 0, stream>>>(
        hbuf, gAS[l], gAD[l], a_s, a_d, amax, den);
    ce_kernel<<<1, NHEADS, 0, stream>>>(gWE[l], gAE[l], cebuf);
    alpha_kernel<<<(nTot * NHEADS + 255) / 256, 256, 0, stream>>>(
        a_s, a_d, eattr, cebuf, esum, ecnt, srcI, dstI, exbuf, amax, nE, nTot);
    exps_kernel<<<(nTot * NHEADS + 255) / 256, 256, 0, stream>>>(
        exbuf, amax, dstI, den, nE, nTot);

    // out = bias; out[dst] += h[src] * coef; relu; f16 for next layer
    binit_kernel<<<(totF + 255) / 256, 256, 0, stream>>>(gB[l], obuf, totF);
    aggregate_kernel<<<(nTot + 7) / 8, 256, 0, stream>>>(
        hbuf, exbuf, den, srcI, dstI, obuf, nE, nTot);
    relu_cvt_kernel<<<(totF + 255) / 256, 256, 0, stream>>>(obuf, act16, totF);
  }

  // ---- 4. pooling + comb + fc1 (WMMA) + fc2 heads ----
  pool_kernel<<<(NB * FDIM + 255) / 256, 256, 0, stream>>>(obuf, gemb);
  comb_kernel<<<(NB * NAG * 1024 + 255) / 256, 256, 0, stream>>>(obuf, gemb, comb16);

  cvt_wt<<<(256 * 1024 + 255) / 256, 256, 0, stream>>>(fc1w, wt16, 1024, 256, 1024);
  {
    int blocks = ((NB * NAG) >> 7) * (256 >> 4);  // 10 * 16
    gemm_wmma_lds<<<blocks, 256, 0, stream>>>(
        comb16, wt16, fc1b, hh, nullptr, NB * NAG, 256, 1024, 1);
  }
  head_kernel<<<(NB * NAG * 6 + 127) / 128, 128, 0, stream>>>(
      hh, fc2w, fc2b, out_mean, out_std);
}
